// Multi_DefectModel_NOGAT_22986664968813
// MI455X (gfx1250) — matrix operand order repaired and checked by reading, not compiled, not run on hardware
//
#include <hip/hip_runtime.h>
#include <hip/hip_bf16.h>
#include <math.h>

// ---------------------------------------------------------------------------
// Multi_DefectModel_NOGAT on MI455X (gfx1250, wave32, WMMA)
// B=256, N=100, C=512, NBLK=8
// ---------------------------------------------------------------------------

typedef __attribute__((ext_vector_type(16))) _Float16 v16h;
typedef __attribute__((ext_vector_type(4)))  _Float16 v4h;
typedef __attribute__((ext_vector_type(8)))  float    v8f;

#define BATCH   256
#define NNODE   100
#define CCH     512
#define NBLK    8
#define EPS_BN  1e-5f

__device__ __forceinline__ v8f zero8() {
  v8f z = {0.f,0.f,0.f,0.f,0.f,0.f,0.f,0.f};
  return z;
}

// A fragment: 16(M) x 32(K) f16 tile, row-major in LDS, leading dim `ld`.
// ISA 7.12.2: lane holds row M=l&15; halves = two contiguous 8-half chunks at
// K = hi*8 and K = 16+hi*8  (compiler lowers to 2x ds_load_b128).
__device__ __forceinline__ v16h load_a_frag(const _Float16* __restrict__ t, int ld, int lane) {
  int m  = lane & 15;
  int hi = (lane >> 4) & 1;
  const _Float16* row = t + m * ld;
  v16h a;
#pragma unroll
  for (int r = 0; r < 8; ++r) {
    int k = ((r >= 4) ? 16 : 0) + hi * 8 + 2 * (r & 3);
    a[2 * r]     = row[k];
    a[2 * r + 1] = row[k + 1];
  }
  return a;
}

// B fragment for D = A(16x32) * B(32x16): B stored TRANSPOSED in LDS as
// N(16 rows) x K(32 cols) row-major. lane holds col n=l&15; two 8-half chunks
// at K = hi*8 and K = 16+hi*8 (2x ds_load_b128), matching the A fragment.
__device__ __forceinline__ v16h load_b_frag(const _Float16* __restrict__ t, int ld, int lane) {
  int n  = lane & 15;
  int kb = (lane < 16) ? 0 : 8;
  const _Float16* row = t + n * ld;
  v16h b;
#pragma unroll
  for (int r = 0; r < 8; ++r) {
    int k = ((r >= 4) ? 16 : 0) + kb + 2 * (r & 3);
    b[2 * r]     = row[k];
    b[2 * r + 1] = row[k + 1];
  }
  return b;
}

__device__ __forceinline__ float elu_f(float x) {
  return x > 0.f ? x : (expf(x) - 1.f);
}

// ---------------------------------------------------------------------------
// BN stats -> per-column affine (scale, shift), 4 columns per block,
// float4 (b128) row loads. Requires K % 4 == 0.
// ---------------------------------------------------------------------------
__global__ void bn_col_affine4(const float* __restrict__ X, int M, int K,
                               const float* __restrict__ g, const float* __restrict__ bta,
                               float* __restrict__ scale, float* __restrict__ shift) {
  __shared__ float s1[4][256];
  __shared__ float s2[4][256];
  int col4 = blockIdx.x * 4;
  float a0=0,a1=0,a2=0,a3=0,q0=0,q1=0,q2=0,q3=0;
  for (int r = threadIdx.x; r < M; r += 256) {
    float4 v = *(const float4*)(X + (long)r * K + col4);
    a0+=v.x; q0+=v.x*v.x; a1+=v.y; q1+=v.y*v.y;
    a2+=v.z; q2+=v.z*v.z; a3+=v.w; q3+=v.w*v.w;
  }
  s1[0][threadIdx.x]=a0; s1[1][threadIdx.x]=a1; s1[2][threadIdx.x]=a2; s1[3][threadIdx.x]=a3;
  s2[0][threadIdx.x]=q0; s2[1][threadIdx.x]=q1; s2[2][threadIdx.x]=q2; s2[3][threadIdx.x]=q3;
  __syncthreads();
  for (int off = 128; off > 0; off >>= 1) {
    if (threadIdx.x < off) {
#pragma unroll
      for (int c = 0; c < 4; ++c) {
        s1[c][threadIdx.x] += s1[c][threadIdx.x + off];
        s2[c][threadIdx.x] += s2[c][threadIdx.x + off];
      }
    }
    __syncthreads();
  }
  if (threadIdx.x < 4) {
    int c = threadIdx.x, col = col4 + c;
    float mean = s1[c][0] / (float)M;
    float var  = s2[c][0] / (float)M - mean * mean;
    float sc   = g[col] * rsqrtf(var + EPS_BN);
    scale[col] = sc;
    shift[col] = bta[col] - mean * sc;
  }
}

// ---------------------------------------------------------------------------
// BatchNorm1d over channel axis for (Bn, N, L): stats over (batch, L) per n.
// One block per n; float4 loads (L % 4 == 0: 768 and 4).
// ---------------------------------------------------------------------------
__global__ void bn_node_affine(const float* __restrict__ X, int Bn, int N, int L,
                               const float* __restrict__ g, const float* __restrict__ bta,
                               float* __restrict__ scale, float* __restrict__ shift) {
  __shared__ float s1[256];
  __shared__ float s2[256];
  int n = blockIdx.x;
  int L4 = L >> 2;
  int total4 = Bn * L4;
  float a = 0.f, q = 0.f;
  for (int e = threadIdx.x; e < total4; e += 256) {
    int b = e / L4, l4 = (e - b * L4) << 2;
    float4 v = *(const float4*)(X + ((long)b * N + n) * L + l4);
    a += v.x + v.y + v.z + v.w;
    q += v.x*v.x + v.y*v.y + v.z*v.z + v.w*v.w;
  }
  s1[threadIdx.x] = a; s2[threadIdx.x] = q;
  __syncthreads();
  for (int off = 128; off > 0; off >>= 1) {
    if (threadIdx.x < off) {
      s1[threadIdx.x] += s1[threadIdx.x + off];
      s2[threadIdx.x] += s2[threadIdx.x + off];
    }
    __syncthreads();
  }
  if (threadIdx.x == 0) {
    int total = Bn * L;
    float mean = s1[0] / (float)total;
    float var  = s2[0] / (float)total - mean * mean;
    float sc   = g[n] * rsqrtf(var + EPS_BN);
    scale[n] = sc;
    shift[n] = bta[n] - mean * sc;
  }
}

// Expand per-node affine to per-row (removes % from the GEMM hot loop).
__global__ void expand_rows(const float* __restrict__ sN, const float* __restrict__ hN,
                            float* __restrict__ sM, float* __restrict__ hM,
                            int mod, long M) {
  long i = (long)blockIdx.x * 256 + threadIdx.x;
  if (i >= M) return;
  int n = (int)(i % mod);
  sM[i] = sN[n];
  hM[i] = hN[n];
}

// ---------------------------------------------------------------------------
// WMMA NT GEMM:  C[M x Nn] = act( A'[M x K] * Bw[Nn x K]^T + bias )
// AFFINE: 0 = none, 1 = per-column (colScale/colShift, len K),
//         2 = per-row (rowScale/rowShift, len M, pre-expanded).
// Requires: M % 128 == 0, K % 32 == 0. Only the N edge is guarded.
// 256 threads = 8 waves (4x2), block tile 128x64, wave tile 32x32.
// ---------------------------------------------------------------------------
template <int AFFINE, int ELU>
__global__ void __launch_bounds__(256)
wmma_gemm_nt(const float* __restrict__ A, int lda, int M, int K,
             const float* __restrict__ Bw,              // Nn x K row-major
             const float* __restrict__ bias,            // length Nn
             float* __restrict__ C, int ldc, int Nn,
             const float* __restrict__ colScale, const float* __restrict__ colShift,
             const float* __restrict__ rowScale, const float* __restrict__ rowShift) {
  __shared__ _Float16 As[128 * 32];
  __shared__ _Float16 Bs[64 * 32];

  int tid  = threadIdx.x;
  int lane = tid & 31;
  int w    = tid >> 5;
  int wrow = w >> 1;            // 0..3
  int wcol = w & 1;             // 0..1
  int m0 = blockIdx.y * 128;
  int n0 = blockIdx.x * 64;

  v8f acc00 = zero8(), acc01 = zero8(), acc10 = zero8(), acc11 = zero8();

  for (int k0 = 0; k0 < K; k0 += 32) {
    // ---- stage A tile (128 x 32) as float4 -> f16, BN affine fused ----
#pragma unroll
    for (int e = tid; e < 1024; e += 256) {        // 8 float4 per row
      int r  = e >> 3;
      int c4 = (e & 7) << 2;
      const float* ap = A + (long)(m0 + r) * lda + k0 + c4;
      float4 v = *(const float4*)ap;
      if (AFFINE == 1) {
        float4 cs = *(const float4*)(colScale + k0 + c4);
        float4 ch = *(const float4*)(colShift + k0 + c4);
        v.x = v.x * cs.x + ch.x; v.y = v.y * cs.y + ch.y;
        v.z = v.z * cs.z + ch.z; v.w = v.w * cs.w + ch.w;
      } else if (AFFINE == 2) {
        float s = rowScale[m0 + r], h = rowShift[m0 + r];
        v.x = v.x * s + h; v.y = v.y * s + h; v.z = v.z * s + h; v.w = v.w * s + h;
      }
      if (c4 == 0 && k0 + 32 < K) __builtin_prefetch(ap + 32, 0, 1);  // next 128B slab line
      v4h hv = {(_Float16)v.x, (_Float16)v.y, (_Float16)v.z, (_Float16)v.w};
      *(v4h*)(As + (r << 5) + c4) = hv;
    }
    // ---- stage B tile (64 x 32) ----
#pragma unroll
    for (int e = tid; e < 512; e += 256) {
      int r  = e >> 3;
      int c4 = (e & 7) << 2;
      int gr = n0 + r;
      float4 v = {0.f, 0.f, 0.f, 0.f};
      if (gr < Nn) v = *(const float4*)(Bw + (long)gr * K + k0 + c4);
      v4h hv = {(_Float16)v.x, (_Float16)v.y, (_Float16)v.z, (_Float16)v.w};
      *(v4h*)(Bs + (r << 5) + c4) = hv;
    }
    __syncthreads();

    v16h a0 = load_a_frag(As + (wrow * 32 + 0)  * 32, 32, lane);
    v16h a1 = load_a_frag(As + (wrow * 32 + 16) * 32, 32, lane);
    v16h b0 = load_b_frag(Bs + (wcol * 32 + 0)  * 32, 32, lane);
    v16h b1 = load_b_frag(Bs + (wcol * 32 + 16) * 32, 32, lane);

    acc00 = __builtin_amdgcn_wmma_f32_16x16x32_f16(false, a0, false, b0, (short)0, acc00, false, false);
    acc01 = __builtin_amdgcn_wmma_f32_16x16x32_f16(false, a0, false, b1, (short)0, acc01, false, false);
    acc10 = __builtin_amdgcn_wmma_f32_16x16x32_f16(false, a1, false, b0, (short)0, acc10, false, false);
    acc11 = __builtin_amdgcn_wmma_f32_16x16x32_f16(false, a1, false, b1, (short)0, acc11, false, false);
    __syncthreads();
  }

  // ---- epilogue (D layout: VGPR r -> M=r / r+8 by lane half) ----
  int hi = (lane >> 4) & 1;
  int ln = lane & 15;
#pragma unroll
  for (int mi = 0; mi < 2; ++mi) {
#pragma unroll
    for (int ni = 0; ni < 2; ++ni) {
      v8f acc = (mi == 0) ? (ni == 0 ? acc00 : acc01) : (ni == 0 ? acc10 : acc11);
      int col = n0 + wcol * 32 + ni * 16 + ln;
      if (col < Nn) {
        float bv = bias[col];
#pragma unroll
        for (int r = 0; r < 8; ++r) {
          int row = m0 + wrow * 32 + mi * 16 + r + 8 * hi;
          float v = acc[r] + bv;
          if (ELU) v = elu_f(v);
          C[(long)row * ldc + col] = v;
        }
      }
    }
  }
}

// ---------------------------------------------------------------------------
// bbox branch: p = elu(bn_ch(pos) @ fcbbox_w^T + b) -> V columns [480,512)
// ---------------------------------------------------------------------------
__global__ void bbox_kernel(const float* __restrict__ pos,
                            const float* __restrict__ w, const float* __restrict__ bias,
                            const float* __restrict__ sc, const float* __restrict__ sh,
                            float* __restrict__ V) {
  int idx = blockIdx.x * 256 + threadIdx.x;
  if (idx >= BATCH * NNODE * 32) return;
  int row = idx >> 5, o = idx & 31;
  int n = row % NNODE;
  float4 pv = *(const float4*)(pos + row * 4);
  float4 wv = *(const float4*)(w + o * 4);
  float s = sc[n], h = sh[n];
  float acc = bias[o]
            + (pv.x * s + h) * wv.x + (pv.y * s + h) * wv.y
            + (pv.z * s + h) * wv.z + (pv.w * s + h) * wv.w;
  V[(long)row * CCH + 480 + o] = elu_f(acc);
}

// ---------------------------------------------------------------------------
// Fused per-batch non-local attention:
//   R = TH_b @ PH_b^T (100x100, M pad 112, N pad 128 -> uniform 7x8 tile grid,
//       56 tiles = exactly 7 WMMA tiles per wave, no predicates) -> LDS (f32)
//   A = softmax(R, rows) -> LDS (f16)
//   Y_b = A @ G_b (100x512) -> global
// One 256-thread workgroup per batch element; ~107 KB dynamic LDS (320KB WGP).
// ---------------------------------------------------------------------------
__global__ void __launch_bounds__(256)
attention_kernel(const float* __restrict__ TH, const float* __restrict__ PH,
                 const float* __restrict__ G, float* __restrict__ Y) {
  extern __shared__ char smem[];
  _Float16* THs = (_Float16*)smem;              // 112*32 f16
  _Float16* PHs = THs + 112 * 32;               // 128*32 f16 (N padded to 128)
  float*    Rs  = (float*)(PHs + 128 * 32);     // 112*128 f32
  _Float16* Af  = (_Float16*)(Rs + 112 * 128);  // 112*128 f16
  _Float16* Gt  = Af + 112 * 128;               // 128*32 f16 (transposed G tile)

  int b    = blockIdx.x;
  int tid  = threadIdx.x;
  int lane = tid & 31;
  int w    = tid >> 5;
  int hi   = (lane >> 4) & 1;
  int ln   = lane & 15;

  const float* THb = TH + (long)b * NNODE * CCH;
  const float* PHb = PH + (long)b * NNODE * CCH;
  const float* Gb  = G  + (long)b * NNODE * CCH;

  // ---------------- Phase 1: R = TH * PH^T ----------------
  v8f acc[7];
#pragma unroll
  for (int s = 0; s < 7; ++s) acc[s] = zero8();

  for (int k0 = 0; k0 < CCH; k0 += 32) {
    for (int e = tid; e < 896; e += 256) {        // TH: 112 rows x 8 float4
      int r  = e >> 3;
      int c4 = (e & 7) << 2;
      float4 tv = {0.f,0.f,0.f,0.f};
      if (r < NNODE) tv = *(const float4*)(THb + (long)r * CCH + k0 + c4);
      v4h th = {(_Float16)tv.x, (_Float16)tv.y, (_Float16)tv.z, (_Float16)tv.w};
      *(v4h*)(THs + (r << 5) + c4) = th;
    }
#pragma unroll
    for (int e = tid; e < 1024; e += 256) {       // PH: 128 rows x 8 float4
      int r  = e >> 3;
      int c4 = (e & 7) << 2;
      float4 pv = {0.f,0.f,0.f,0.f};
      if (r < NNODE) pv = *(const float4*)(PHb + (long)r * CCH + k0 + c4);
      v4h ph = {(_Float16)pv.x, (_Float16)pv.y, (_Float16)pv.z, (_Float16)pv.w};
      *(v4h*)(PHs + (r << 5) + c4) = ph;
    }
    __syncthreads();
    // uniform 7(M) x 8(N) = 56 tiles -> exactly 7 per wave, no predicates
#pragma unroll
    for (int s = 0; s < 7; ++s) {
      int t = w + 8 * s;
      int i = t >> 3, j = t & 7;
      v16h a  = load_a_frag(THs + i * 16 * 32, 32, lane);
      v16h bf = load_b_frag(PHs + j * 16 * 32, 32, lane);
      acc[s] = __builtin_amdgcn_wmma_f32_16x16x32_f16(false, a, false, bf, (short)0, acc[s], false, false);
    }
    __syncthreads();
  }
#pragma unroll
  for (int s = 0; s < 7; ++s) {
    int t = w + 8 * s;
    int i = t >> 3, j = t & 7;
#pragma unroll
    for (int r = 0; r < 8; ++r)
      Rs[(i * 16 + r + 8 * hi) * 128 + j * 16 + ln] = acc[s][r];
  }
  __syncthreads();

  // ---------------- Phase 2: row softmax -> f16 A matrix ----------------
  if (tid < NNODE) {
    float mx = -1e30f;
    for (int c = 0; c < NNODE; ++c) mx = fmaxf(mx, Rs[tid * 128 + c]);
    float sum = 0.f;
    for (int c = 0; c < NNODE; ++c) {
      float e = expf(Rs[tid * 128 + c] - mx);
      Rs[tid * 128 + c] = e;
      sum += e;
    }
    float inv = 1.f / sum;
    for (int c = 0; c < NNODE; ++c) Af[tid * 128 + c] = (_Float16)(Rs[tid * 128 + c] * inv);
    for (int c = NNODE; c < 128; ++c) Af[tid * 128 + c] = (_Float16)0.f;
  } else if (tid < 112) {
    for (int c = 0; c < 128; ++c) Af[tid * 128 + c] = (_Float16)0.f;
  }
  __syncthreads();

  // ---------------- Phase 3: Y = A * G ----------------
  for (int n0 = 0; n0 < CCH; n0 += 128) {
    v8f acy[7];
#pragma unroll
    for (int s = 0; s < 7; ++s) acy[s] = zero8();

    for (int k0 = 0; k0 < 128; k0 += 32) {
      // stage G[k0..k0+31][n0..n0+127] transposed into Gt (N x K layout)
      for (int e = tid; e < 1024; e += 256) {     // 32 rows x 32 float4
        int kk = e >> 5;
        int n4 = (e & 31) << 2;
        int gr = k0 + kk;
        float4 v = {0.f,0.f,0.f,0.f};
        if (gr < NNODE) v = *(const float4*)(Gb + (long)gr * CCH + n0 + n4);
        Gt[(n4 + 0) * 32 + kk] = (_Float16)v.x;
        Gt[(n4 + 1) * 32 + kk] = (_Float16)v.y;
        Gt[(n4 + 2) * 32 + kk] = (_Float16)v.z;
        Gt[(n4 + 3) * 32 + kk] = (_Float16)v.w;
      }
      __syncthreads();
#pragma unroll
      for (int s = 0; s < 7; ++s) {               // 7x8 = 56 tiles, 7 per wave
        int t = w + 8 * s;
        int i = t >> 3, j = t & 7;
        v16h a  = load_a_frag(Af + i * 16 * 128 + k0, 128, lane);
        v16h bf = load_b_frag(Gt + j * 16 * 32, 32, lane);
        acy[s] = __builtin_amdgcn_wmma_f32_16x16x32_f16(false, a, false, bf, (short)0, acy[s], false, false);
      }
      __syncthreads();
    }
#pragma unroll
    for (int s = 0; s < 7; ++s) {
      int t = w + 8 * s;
      int i = t >> 3, j = t & 7;
#pragma unroll
      for (int r = 0; r < 8; ++r) {
        int m = i * 16 + r + 8 * hi;
        if (m < NNODE)
          Y[((long)b * NNODE + m) * CCH + n0 + j * 16 + ln] = acy[s][r];
      }
    }
    __syncthreads();
  }
}

// ---------------------------------------------------------------------------
// V = V + BN(WY), float4 vectorized (CCH % 4 == 0)
// ---------------------------------------------------------------------------
__global__ void bn_residual4(const float* __restrict__ WY,
                             const float* __restrict__ sc, const float* __restrict__ sh,
                             float* __restrict__ V, long total4) {
  long i4 = (long)blockIdx.x * 256 + threadIdx.x;
  if (i4 >= total4) return;
  long idx = i4 << 2;
  int c = (int)(idx & (CCH - 1));
  float4 wv = *(const float4*)(WY + idx);
  float4 vv = *(const float4*)(V + idx);
  float4 s  = *(const float4*)(sc + c);
  float4 h  = *(const float4*)(sh + c);
  vv.x += wv.x * s.x + h.x;
  vv.y += wv.y * s.y + h.y;
  vv.z += wv.z * s.z + h.z;
  vv.w += wv.w * s.w + h.w;
  *(float4*)(V + idx) = vv;
}

// ---------------------------------------------------------------------------
// h_feature[b,c] = sum_n V[b,n,c] / (N * sqrt(sum_n V[b,n,c]^2))
// ---------------------------------------------------------------------------
__global__ void l2_mean(const float* __restrict__ V, float* __restrict__ allf) {
  int idx = blockIdx.x * 256 + threadIdx.x;
  if (idx >= BATCH * CCH) return;
  int b = idx >> 9, c = idx & (CCH - 1);
  float s = 0.f, ss = 0.f;
  for (int n = 0; n < NNODE; ++n) {
    float v = V[((long)b * NNODE + n) * CCH + c];
    s += v; ss += v * v;
  }
  allf[(long)b * 1536 + 512 + c] = s / ((float)NNODE * sqrtf(ss));
}

// ---------------------------------------------------------------------------
// out = BN(allf) @ ff_w^T + ff_b  (256 x 2), float4 dot loop
// ---------------------------------------------------------------------------
__global__ void final_out(const float* __restrict__ allf,
                          const float* __restrict__ sc, const float* __restrict__ sh,
                          const float* __restrict__ w, const float* __restrict__ bias,
                          float* __restrict__ out) {
  int idx = blockIdx.x * 256 + threadIdx.x;
  if (idx >= BATCH * 2) return;
  int b = idx >> 1, o = idx & 1;
  float acc = bias[o];
  for (int k = 0; k < 1536; k += 4) {
    float4 a = *(const float4*)(allf + (long)b * 1536 + k);
    float4 s = *(const float4*)(sc + k);
    float4 h = *(const float4*)(sh + k);
    float4 wv = *(const float4*)(w + o * 1536 + k);
    acc += (a.x * s.x + h.x) * wv.x + (a.y * s.y + h.y) * wv.y
         + (a.z * s.z + h.z) * wv.z + (a.w * s.w + h.w) * wv.w;
  }
  out[b * 2 + o] = acc;
}

// ---------------------------------------------------------------------------
extern "C" void kernel_launch(void* const* d_in, const int* in_sizes, int n_in,
                              void* d_out, int out_size, void* d_ws, size_t ws_size,
                              hipStream_t stream) {
  const float* img      = (const float*)d_in[0];
  const float* txt      = (const float*)d_in[1];
  const float* node     = (const float*)d_in[2];
  const float* pos      = (const float*)d_in[3];
  const float* swinbn_g = (const float*)d_in[4];
  const float* swinbn_b = (const float*)d_in[5];
  const float* swinfc_w = (const float*)d_in[6];
  const float* swinfc_b = (const float*)d_in[7];
  const float* bntext_g = (const float*)d_in[8];
  const float* bntext_b = (const float*)d_in[9];
  const float* fctext_w = (const float*)d_in[10];
  const float* fctext_b = (const float*)d_in[11];
  const float* bngat_g  = (const float*)d_in[12];
  const float* bngat_b  = (const float*)d_in[13];
  const float* fcgat_w  = (const float*)d_in[14];
  const float* fcgat_b  = (const float*)d_in[15];
  const float* bnbbox_g = (const float*)d_in[16];
  const float* bnbbox_b = (const float*)d_in[17];
  const float* fcbbox_w = (const float*)d_in[18];
  const float* fcbbox_b = (const float*)d_in[19];
  const float* th_w     = (const float*)d_in[20];
  const float* th_b     = (const float*)d_in[21];
  const float* ph_w     = (const float*)d_in[22];
  const float* ph_b     = (const float*)d_in[23];
  const float* g_w      = (const float*)d_in[24];
  const float* g_b      = (const float*)d_in[25];
  const float* W_w      = (const float*)d_in[26];
  const float* W_b      = (const float*)d_in[27];
  const float* bn_g     = (const float*)d_in[28];
  const float* bn_b     = (const float*)d_in[29];
  const float* ffbn_g   = (const float*)d_in[30];
  const float* ffbn_b   = (const float*)d_in[31];
  const float* ff_w     = (const float*)d_in[32];
  const float* ff_b     = (const float*)d_in[33];
  float* out = (float*)d_out;

  // ---- workspace carve (floats) ----
  const long MR = (long)BATCH * NNODE;       // 25600 rows
  const long MC = MR * CCH;                  // 13,107,200 floats per big buffer
  float* ws  = (float*)d_ws;
  float* V   = ws;
  float* G   = V  + MC;
  float* TH  = G  + MC;
  float* PH  = TH + MC;
  float* Y   = PH + MC;
  float* WY  = TH;                           // TH dead after attention -> reuse
  float* allf = Y + MC;                      // 256 x 1536
  float* p = allf + (long)BATCH * 1536;
  float* imgS  = p; p += 1024;  float* imgH  = p; p += 1024;
  float* txtS  = p; p += 768;   float* txtH  = p; p += 768;
  float* nodeS = p; p += NNODE; float* nodeH = p; p += NNODE;
  float* bbS   = p; p += NNODE; float* bbH   = p; p += NNODE;
  float* wyS   = p; p += CCH;   float* wyH   = p; p += CCH;
  float* fS    = p; p += 1536;  float* fH    = p; p += 1536;
  float* rowS  = p; p += MR;    float* rowH  = p; p += MR;   // expanded node affine

  const size_t ATT_SMEM = (size_t)(112 * 32 + 128 * 32 + 112 * 128 + 128 * 32) * sizeof(_Float16)
                        + (size_t)(112 * 128) * sizeof(float);   // 109,568 B

  // ---- BN stats for input branches ----
  bn_col_affine4<<<256, 256, 0, stream>>>(img, BATCH, 1024, swinbn_g, swinbn_b, imgS, imgH);
  bn_col_affine4<<<192, 256, 0, stream>>>(txt, BATCH, 768,  bntext_g, bntext_b, txtS, txtH);
  bn_node_affine<<<NNODE, 256, 0, stream>>>(node, BATCH, NNODE, 768, bngat_g, bngat_b, nodeS, nodeH);
  bn_node_affine<<<NNODE, 256, 0, stream>>>(pos,  BATCH, NNODE, 4,   bnbbox_g, bnbbox_b, bbS, bbH);
  expand_rows<<<(int)((MR + 255) / 256), 256, 0, stream>>>(nodeS, nodeH, rowS, rowH, NNODE, MR);

  // ---- branch GEMMs (BN fused into A-load, ELU epilogue) ----
  wmma_gemm_nt<1, 1><<<dim3(8, 2), 256, 0, stream>>>(img, 1024, BATCH, 1024, swinfc_w, swinfc_b,
                                                     allf, 1536, 512, imgS, imgH, nullptr, nullptr);
  wmma_gemm_nt<1, 1><<<dim3(8, 2), 256, 0, stream>>>(txt, 768, BATCH, 768, fctext_w, fctext_b,
                                                     allf + 1024, 1536, 512, txtS, txtH, nullptr, nullptr);
  wmma_gemm_nt<2, 1><<<dim3(8, 200), 256, 0, stream>>>(node, 768, (int)MR, 768, fcgat_w, fcgat_b,
                                                       V, CCH, 480, nullptr, nullptr, rowS, rowH);
  bbox_kernel<<<(BATCH * NNODE * 32) / 256, 256, 0, stream>>>(pos, fcbbox_w, fcbbox_b, bbS, bbH, V);

  // ---- 8 Rs_GCN non-local blocks ----
  const long WSTEP = (long)CCH * CCH;  // 262144
  for (int i = 0; i < NBLK; ++i) {
    wmma_gemm_nt<0, 0><<<dim3(8, 200), 256, 0, stream>>>(V, CCH, (int)MR, CCH, g_w + i * WSTEP, g_b + i * CCH,
                                                         G, CCH, CCH, nullptr, nullptr, nullptr, nullptr);
    wmma_gemm_nt<0, 0><<<dim3(8, 200), 256, 0, stream>>>(V, CCH, (int)MR, CCH, th_w + i * WSTEP, th_b + i * CCH,
                                                         TH, CCH, CCH, nullptr, nullptr, nullptr, nullptr);
    wmma_gemm_nt<0, 0><<<dim3(8, 200), 256, 0, stream>>>(V, CCH, (int)MR, CCH, ph_w + i * WSTEP, ph_b + i * CCH,
                                                         PH, CCH, CCH, nullptr, nullptr, nullptr, nullptr);
    attention_kernel<<<BATCH, 256, ATT_SMEM, stream>>>(TH, PH, G, Y);
    wmma_gemm_nt<0, 0><<<dim3(8, 200), 256, 0, stream>>>(Y, CCH, (int)MR, CCH, W_w + i * WSTEP, W_b + i * CCH,
                                                         WY, CCH, CCH, nullptr, nullptr, nullptr, nullptr);
    bn_col_affine4<<<CCH / 4, 256, 0, stream>>>(WY, (int)MR, CCH, bn_g + i * CCH, bn_b + i * CCH, wyS, wyH);
    bn_residual4<<<(int)(MC / 4 / 256), 256, 0, stream>>>(WY, wyS, wyH, V, MC / 4);
  }

  // ---- l2norm over nodes + mean -> h_feature -> allf[:, 512:1024] ----
  l2_mean<<<(BATCH * CCH) / 256, 256, 0, stream>>>(V, allf);

  // ---- final classifier ----
  bn_col_affine4<<<1536 / 4, 256, 0, stream>>>(allf, BATCH, 1536, ffbn_g, ffbn_b, fS, fH);
  final_out<<<2, 256, 0, stream>>>(allf, fS, fH, ff_w, ff_b, out);
}
